// PETAttention_12970801233963
// MI455X (gfx1250) — matrix operand order repaired and checked by reading, not compiled, not run on hardware
//
#include <hip/hip_runtime.h>
#include <hip/hip_bf16.h>
#include <math.h>

// PET attention, fully fused per batch on MI455X (gfx1250, wave32, WMMA).
// One workgroup (256 threads = 8 waves) processes one batch element b:
//   LDS:  x(bf16) -> qkv GEMM -> q,k,vT (bf16) -> per-head WMMA attention
//         -> attn out (bf16, aliases x) -> out-proj GEMM -> global f32.
// All GEMMs use v_wmma_f32_16x16x32_bf16 (f32 accumulate).
// Weights are pre-converted fp32->bf16 once per launch into d_ws so the hot
// GEMM loops load bf16 B-fragments directly (no per-tile v_cvt work).

typedef __attribute__((ext_vector_type(16))) __bf16 v16bf;
typedef __attribute__((ext_vector_type(8)))  __bf16 v8bf;
typedef __attribute__((ext_vector_type(4)))  __bf16 v4bf;
typedef __attribute__((ext_vector_type(8)))  float  v8f;

#define SEQ  128
#define DIM  256
#define NH   8
#define DH   32
#define LDX  264   // row stride (elems) for xs/qs/ks/os : 528B -> 16-row b128 reads hit distinct bank groups
#define LVT  136   // row stride for v-transposed [d][s]
#define LPS  136   // row stride for per-wave prob scratch [16][128]

// dynamic-LDS carve (bytes)
#define OFF_XS   0
#define SZ_XS    (SEQ * LDX * 2)          // 67584  (reused as attention-output "os")
#define OFF_QS   (OFF_XS + SZ_XS)
#define OFF_KS   (OFF_QS + SZ_XS)
#define OFF_VT   (OFF_KS + SZ_XS)
#define SZ_VT    (DIM * LVT * 2)          // 69632
#define OFF_PS   (OFF_VT + SZ_VT)
#define SZ_PS    (8 * 16 * LPS * 2)       // 34816
#define OFF_RB   (OFF_PS + SZ_PS)
#define SZ_RB    (SEQ * 4)
#define SMEM_BYTES (OFF_RB + SZ_RB)       // 307712 < 320KB

#define WB_IN_ELEMS  (3 * DIM * DIM)      // 196608
#define WB_OUT_ELEMS (DIM * DIM)          // 65536
#define WS_NEEDED    ((WB_IN_ELEMS + WB_OUT_ELEMS) * 2)  // 524288 B

// --- fragment loaders -------------------------------------------------------
// 16-bit A 16x32 (MxK): lanes0-15: M=lane, K=kb..kb+15;
//                       lanes16-31: M=lane-16, K=kb+16..kb+31.
__device__ inline v16bf load_a_frag(const __bf16* base, int ld, int row0, int kbase) {
  const int lane = threadIdx.x & 31;
  const int m    = row0 + (lane & 15);
  const int koff = kbase + ((lane >> 4) << 4);
  const __bf16* p = base + m * ld + koff;
  v8bf lo = *(const v8bf*)p;          // 16B aligned by construction
  v8bf hi = *(const v8bf*)(p + 8);
  v16bf a;
#pragma unroll
  for (int i = 0; i < 8; ++i) { a[i] = lo[i]; a[i + 8] = hi[i]; }
  return a;
}

// 16-bit B 32x16 (KxN): lanes0-15: N=lane, K=kb..kb+15; lanes16-31: N=lane-16, K=kb+16..kb+31.
// Source stored with N as the row and K contiguous: arr[n*ld + k].
__device__ inline v16bf load_b_frag_lds(const __bf16* base, int ld, int n0, int kbase) {
  const int lane = threadIdx.x & 31;
  const int n = n0 + (lane & 15);
  const int k = kbase + ((lane >> 4) << 4);
  const __bf16* p = base + n * ld + k;   // 16B aligned by construction
  v8bf lo = *(const v8bf*)p;
  v8bf hi = *(const v8bf*)(p + 8);
  v16bf b;
#pragma unroll
  for (int i = 0; i < 8; ++i) { b[i] = lo[i]; b[i + 8] = hi[i]; }
  return b;
}

// B fragment from pre-converted bf16 weight [N][K] in global (L2-resident):
// each lane reads 16 contiguous bf16 (32B).
__device__ inline v16bf load_b_frag_wb(const __bf16* __restrict__ W, int ldw, int n0, int kbase) {
  const int lane = threadIdx.x & 31;
  const int n = n0 + (lane & 15);
  const int k = kbase + ((lane >> 4) << 4);
  const v8bf* p = (const v8bf*)(W + (size_t)n * ldw + k);
  v8bf lo = p[0], hi = p[1];
  v16bf b;
#pragma unroll
  for (int i = 0; i < 8; ++i) { b[i] = lo[i]; b[i + 8] = hi[i]; }
  return b;
}

// Fallback: B fragment streamed from row-major f32 weight (converted in-flight).
__device__ inline v16bf load_b_frag_wf(const float* __restrict__ W, int ldw, int n0, int kbase) {
  const int lane = threadIdx.x & 31;
  const int n = n0 + (lane & 15);
  const int k = kbase + ((lane >> 4) << 4);
  const float4* p = (const float4*)(W + (size_t)n * ldw + k);
  float4 c0 = p[0], c1 = p[1], c2 = p[2], c3 = p[3];
  v16bf b;
  b[0]=(__bf16)c0.x;  b[1]=(__bf16)c0.y;  b[2]=(__bf16)c0.z;  b[3]=(__bf16)c0.w;
  b[4]=(__bf16)c1.x;  b[5]=(__bf16)c1.y;  b[6]=(__bf16)c1.z;  b[7]=(__bf16)c1.w;
  b[8]=(__bf16)c2.x;  b[9]=(__bf16)c2.y;  b[10]=(__bf16)c2.z; b[11]=(__bf16)c2.w;
  b[12]=(__bf16)c3.x; b[13]=(__bf16)c3.y; b[14]=(__bf16)c3.z; b[15]=(__bf16)c3.w;
  return b;
}

template<bool WB>
__device__ inline v16bf load_b_frag_w(const void* __restrict__ W, int ldw, int n0, int kbase) {
  if (WB) return load_b_frag_wb((const __bf16*)W, ldw, n0, kbase);
  else    return load_b_frag_wf((const float*)W, ldw, n0, kbase);
}

__device__ inline v8f wmma_bf16(v16bf a, v16bf b, v8f c) {
  return __builtin_amdgcn_wmma_f32_16x16x32_bf16(false, a, false, b, (short)0, c, false, false);
}

// ---- weight pre-conversion: fp32 -> bf16 into d_ws (runs once per launch) ---
__global__ __launch_bounds__(256)
void convert_w_kernel(const float* __restrict__ W_in, const float* __restrict__ W_out,
                      __bf16* __restrict__ wb_in, __bf16* __restrict__ wb_out)
{
  const int i = blockIdx.x * 256 + threadIdx.x;       // 65536 threads, 1 float4 each
  if (i < WB_IN_ELEMS / 4) {
    float4 v = ((const float4*)W_in)[i];
    v4bf o; o[0]=(__bf16)v.x; o[1]=(__bf16)v.y; o[2]=(__bf16)v.z; o[3]=(__bf16)v.w;
    *(v4bf*)(wb_in + 4 * (size_t)i) = o;
  } else {
    const int j = i - WB_IN_ELEMS / 4;
    float4 v = ((const float4*)W_out)[j];
    v4bf o; o[0]=(__bf16)v.x; o[1]=(__bf16)v.y; o[2]=(__bf16)v.z; o[3]=(__bf16)v.w;
    *(v4bf*)(wb_out + 4 * (size_t)j) = o;
  }
}

template<bool WB>
__global__ __launch_bounds__(256)
void pet_attn_kernel(const float* __restrict__ x,
                     const float* __restrict__ disp,
                     const unsigned char* __restrict__ mask_exists, // jnp bool_ = 1 byte/elem
                     const void* __restrict__ W_in,                 // bf16 if WB else f32, [768][256]
                     const float* __restrict__ b_in,
                     const void* __restrict__ W_out,                // bf16 if WB else f32, [256][256]
                     const float* __restrict__ b_out,
                     float* __restrict__ out)
{
  extern __shared__ __align__(16) char smem[];
  __bf16* xs = (__bf16*)(smem + OFF_XS);  // input x (bf16); later reused as attn output "os"
  __bf16* qs = (__bf16*)(smem + OFF_QS);
  __bf16* ks = (__bf16*)(smem + OFF_KS);
  __bf16* vT = (__bf16*)(smem + OFF_VT);  // v transposed: vT[d][s]
  __bf16* ps = (__bf16*)(smem + OFF_PS);  // per-wave prob tile [16][128]
  float*  rb = (float* )(smem + OFF_RB);  // per-row mask bias

  const int tid  = threadIdx.x;
  const int lane = tid & 31;
  const int wave = tid >> 5;
  const int bidx = blockIdx.x;
  const int row0 = wave * 16;
  const int rlo  = row0 + ((lane >> 4) << 3);  // C/D layout: lanes16-31 hold rows +8
  const int wsz  = WB ? 2 : 4;                 // weight element size (for prefetch addressing)

  // ---- phase 0: stage x -> bf16 LDS; per-row cutoff bias --------------------
  const float* xb = x + (size_t)bidx * SEQ * DIM;
  for (int i = tid; i < SEQ * DIM / 4; i += 256) {
    const int row = i >> 6;             // 64 float4 per row
    const int c4  = (i & 63) << 2;
    float4 v = ((const float4*)xb)[i];
    __bf16* d = xs + row * LDX + c4;
    d[0]=(__bf16)v.x; d[1]=(__bf16)v.y; d[2]=(__bf16)v.z; d[3]=(__bf16)v.w;
  }
  if (tid < SEQ) {
    const size_t ri = (size_t)bidx * SEQ + tid;
    const float dx = disp[ri*3+0], dy = disp[ri*3+1], dz = disp[ri*3+2];
    const float len = sqrtf(dx*dx + dy*dy + dz*dz);
    float bias = 0.0f;                                  // r_cut=3, delta=1 -> band (2,3)
    if (len > 2.0f && len < 3.0f) {
      const float y = len - 2.0f;
      bias = -__expf(-1.0f / (y * y)) / (1.0f - y);
    }
    if (!mask_exists[ri]) bias = -INFINITY;
    rb[tid] = bias;
  }
  __syncthreads();

  // ---- phase 1: qkv = x @ W_in^T + b_in  (wave owns 16 rows, all 768 cols) --
  {
    v16bf afr[8];
#pragma unroll
    for (int kt = 0; kt < 8; ++kt) afr[kt] = load_a_frag(xs, LDX, row0, kt * 32);
    for (int nt = 0; nt < 48; ++nt) {
      if (nt + 1 < 48)  // pull next weight tile toward L2/L0 (global_prefetch_b8)
        __builtin_prefetch((const char*)W_in + ((size_t)(nt + 1) * 16 * DIM + lane * 128) * wsz, 0, 1);
      v8f acc = {};
#pragma unroll
      for (int kt = 0; kt < 8; ++kt)
        acc = wmma_bf16(afr[kt], load_b_frag_w<WB>(W_in, DIM, nt * 16, kt * 32), acc);
      const int nc = nt * 16 + (lane & 15);
      const float bias = b_in[nc];
      if (nc < 256) {              // q : row-major [s][d]
#pragma unroll
        for (int r = 0; r < 8; ++r) qs[(rlo + r) * LDX + nc] = (__bf16)(acc[r] + bias);
      } else if (nc < 512) {       // k : row-major [s][d]
#pragma unroll
        for (int r = 0; r < 8; ++r) ks[(rlo + r) * LDX + (nc - 256)] = (__bf16)(acc[r] + bias);
      } else {                     // v : transposed [d][s] so P@V B-frags are contiguous
#pragma unroll
        for (int r = 0; r < 8; ++r) vT[(nc - 512) * LVT + (rlo + r)] = (__bf16)(acc[r] + bias);
      }
    }
  }
  __syncthreads();

  // ---- phase 2: per-head attention -----------------------------------------
  {
    const float scale = 0.17677669529663687f;  // 1/sqrt(32)
    float rbr[8];
#pragma unroll
    for (int r = 0; r < 8; ++r) rbr[r] = rb[rlo + r];
    __bf16* pw = ps + wave * 16 * LPS;

    for (int h = 0; h < NH; ++h) {
      const int dc = h * DH;
      const v16bf aq = load_a_frag(qs, LDX, row0, dc);   // K-chunk = whole head dim (32)
      v8f sc[8];
#pragma unroll
      for (int jt = 0; jt < 8; ++jt) {                   // scores: q @ k^T
        v8f z = {};
        sc[jt] = wmma_bf16(aq, load_b_frag_lds(ks, LDX, jt * 16, dc), z);
      }
      // scale + row bias + softmax over keys (cols = lanes within 16-lane halves)
#pragma unroll
      for (int r = 0; r < 8; ++r) {
        float m = -INFINITY;
#pragma unroll
        for (int jt = 0; jt < 8; ++jt) {
          sc[jt][r] = sc[jt][r] * scale + rbr[r];
          m = fmaxf(m, sc[jt][r]);
        }
#pragma unroll
        for (int d = 1; d < 16; d <<= 1) m = fmaxf(m, __shfl_xor(m, d, 32));
        float s = 0.0f;
#pragma unroll
        for (int jt = 0; jt < 8; ++jt) { const float e = __expf(sc[jt][r] - m); sc[jt][r] = e; s += e; }
#pragma unroll
        for (int d = 1; d < 16; d <<= 1) s += __shfl_xor(s, d, 32);
        const float inv = 1.0f / s;
#pragma unroll
        for (int jt = 0; jt < 8; ++jt)   // C-layout -> A-layout bounce via LDS (bf16)
          pw[(((lane >> 4) << 3) + r) * LPS + jt * 16 + (lane & 15)] = (__bf16)(sc[jt][r] * inv);
      }
      // attn @ v  (K = seq, 4 chunks of 32; two 16-col D tiles cover DH=32)
      v8f o0 = {}, o1 = {};
#pragma unroll
      for (int jt = 0; jt < 4; ++jt) {
        const v16bf ap = load_a_frag(pw, LPS, 0, jt * 32);
        o0 = wmma_bf16(ap, load_b_frag_lds(vT, LVT, dc,      jt * 32), o0);
        o1 = wmma_bf16(ap, load_b_frag_lds(vT, LVT, dc + 16, jt * 32), o1);
      }
#pragma unroll
      for (int r = 0; r < 8; ++r) {      // attn output -> "os" (aliases xs, consumed already)
        xs[(rlo + r) * LDX + dc +      (lane & 15)] = (__bf16)o0[r];
        xs[(rlo + r) * LDX + dc + 16 + (lane & 15)] = (__bf16)o1[r];
      }
    }
  }
  __syncthreads();

  // ---- phase 3: out = os @ W_out^T + b_out ---------------------------------
  {
    v16bf afr[8];
#pragma unroll
    for (int kt = 0; kt < 8; ++kt) afr[kt] = load_a_frag(xs, LDX, row0, kt * 32);
    float* ob = out + (size_t)bidx * SEQ * DIM;
    for (int nt = 0; nt < 16; ++nt) {
      if (nt + 1 < 16)
        __builtin_prefetch((const char*)W_out + ((size_t)(nt + 1) * 16 * DIM + lane * 128) * wsz, 0, 1);
      v8f acc = {};
#pragma unroll
      for (int kt = 0; kt < 8; ++kt)
        acc = wmma_bf16(afr[kt], load_b_frag_w<WB>(W_out, DIM, nt * 16, kt * 32), acc);
      const int nc = nt * 16 + (lane & 15);
      const float bias = b_out[nc];
#pragma unroll
      for (int r = 0; r < 8; ++r)
        ob[(size_t)(rlo + r) * DIM + nc] = acc[r] + bias;
    }
  }
}

extern "C" void kernel_launch(void* const* d_in, const int* in_sizes, int n_in,
                              void* d_out, int out_size, void* d_ws, size_t ws_size,
                              hipStream_t stream) {
  const float*         x     = (const float*)d_in[0];
  const float*         disp  = (const float*)d_in[1];
  const unsigned char* mask  = (const unsigned char*)d_in[2]; // jnp bool_: 1 byte/elem
  const float*         W_in  = (const float*)d_in[3];
  const float*         b_in  = (const float*)d_in[4];
  const float*         W_out = (const float*)d_in[5];
  const float*         b_out = (const float*)d_in[6];
  float*               out   = (float*)d_out;
  (void)in_sizes; (void)n_in; (void)out_size;

  if (ws_size >= (size_t)WS_NEEDED) {
    __bf16* wb_in  = (__bf16*)d_ws;
    __bf16* wb_out = wb_in + WB_IN_ELEMS;
    convert_w_kernel<<<(WB_IN_ELEMS + WB_OUT_ELEMS) / 4 / 256, 256, 0, stream>>>(
        W_in, W_out, wb_in, wb_out);
    (void)hipFuncSetAttribute(reinterpret_cast<const void*>(pet_attn_kernel<true>),
                              hipFuncAttributeMaxDynamicSharedMemorySize, SMEM_BYTES);
    pet_attn_kernel<true><<<1024, 256, SMEM_BYTES, stream>>>(
        x, disp, mask, (const void*)wb_in, b_in, (const void*)wb_out, b_out, out);
  } else {
    (void)hipFuncSetAttribute(reinterpret_cast<const void*>(pet_attn_kernel<false>),
                              hipFuncAttributeMaxDynamicSharedMemorySize, SMEM_BYTES);
    pet_attn_kernel<false><<<1024, 256, SMEM_BYTES, stream>>>(
        x, disp, mask, (const void*)W_in, b_in, (const void*)W_out, b_out, out);
  }
}
